// Temp_mamba_44487271252004
// MI455X (gfx1250) — matrix operand order repaired and checked by reading, not compiled, not run on hardware
//
#include <hip/hip_runtime.h>
#include <hip/hip_bf16.h>
#include <math.h>

// ---------------- problem constants ----------------
#define Bsz     4
#define Nseq    64
#define Lseq    64
#define DM      128     // d_model
#define DI      256     // d_inner
#define DSTATE  16
#define DCONV   4
#define DTRANK  8
#define NLAYERS 4

typedef __attribute__((ext_vector_type(16))) _Float16 v16h;
typedef __attribute__((ext_vector_type(8)))  _Float16 v8h;
typedef __attribute__((ext_vector_type(8)))  float    v8f;

// ---------------- f16 weight workspace layout (element offsets) ----------------
constexpr int OFF_MLP1 = 0;                          // [128][128]
constexpr int OFF_MLP2 = OFF_MLP1 + DM * DM;         // [128][128]
constexpr int OFF_INW  = OFF_MLP2 + DM * DM;         // [L][512][128]
constexpr int OFF_XPW  = OFF_INW  + NLAYERS * 2 * DI * DM;   // [L][48][256] zero-padded rows 40..47
constexpr int OFF_DTW  = OFF_XPW  + NLAYERS * 48 * DI;       // [L][256][32] zero-padded cols 8..31
constexpr int OFF_OUTW = OFF_DTW  + NLAYERS * DI * 32;       // [L][128][256]
constexpr int TOTAL_H  = OFF_OUTW + NLAYERS * DM * DI;

// ---------------- LDS layout (bytes) ----------------
constexpr int SM_HN16  = 0;                          // f16 [64][128]  rmsnorm'd input / x
constexpr int SM_XC16  = SM_HN16 + 64 * 128 * 2;     // f16 [64][256]  conv out / gated y
constexpr int SM_Z16   = SM_XC16 + 64 * 256 * 2;     // f16 [64][256]  gate z
constexpr int SM_DTIN  = SM_Z16  + 64 * 256 * 2;     // f16 [64][32]   dt_rank padded
constexpr int SM_RESID = SM_DTIN + 64 * 32  * 2;     // f32 [64][128]
constexpr int SM_HBUF  = SM_RESID+ 64 * 128 * 4;     // f32 [64][128]  block output h
constexpr int SM_X     = SM_HBUF + 64 * 128 * 4;     // f32 [64][256]  xc pre-conv -> dt -> y
constexpr int SM_XDBL  = SM_X    + 64 * 256 * 4;     // f32 [64][48]   x_dbl (dt|B|C|pad)
constexpr int SM_SRED  = SM_XDBL + 64 * 48  * 4;     // f32 scratch (scale, valid)
constexpr int SMEM_BYTES = SM_SRED + 64;             // ~224 KB (< 320 KB WGP LDS)

// ---------------- prep: f32 weights -> f16 (padded), A = -exp(A_log) ----------------
__global__ __launch_bounds__(256) void mamba_prep_kernel(
    const float* __restrict__ mlp_w1, const float* __restrict__ mlp_w2,
    const float* __restrict__ in_w,   const float* __restrict__ xp_w,
    const float* __restrict__ dt_w,   const float* __restrict__ out_w,
    const float* __restrict__ A_log,
    _Float16* __restrict__ w16, float* __restrict__ A_ws)
{
    int i = blockIdx.x * blockDim.x + threadIdx.x;
    if (i < DM * DM) { w16[OFF_MLP1 + i] = (_Float16)mlp_w1[i]; return; }
    i -= DM * DM;
    if (i < DM * DM) { w16[OFF_MLP2 + i] = (_Float16)mlp_w2[i]; return; }
    i -= DM * DM;
    if (i < NLAYERS * 2 * DI * DM) { w16[OFF_INW + i] = (_Float16)in_w[i]; return; }
    i -= NLAYERS * 2 * DI * DM;
    if (i < NLAYERS * 48 * DI) {             // x_proj padded: rows 40..47 zero
        int l = i / (48 * DI); int rem = i % (48 * DI);
        int r = rem / DI, c = rem % DI;
        float v = (r < DTRANK + 2 * DSTATE) ? xp_w[(l * 40 + r) * DI + c] : 0.f;
        w16[OFF_XPW + i] = (_Float16)v; return;
    }
    i -= NLAYERS * 48 * DI;
    if (i < NLAYERS * DI * 32) {             // dt_proj padded: cols 8..31 zero
        int l = i / (DI * 32); int rem = i % (DI * 32);
        int r = rem / 32, c = rem % 32;
        float v = (c < DTRANK) ? dt_w[(l * DI + r) * DTRANK + c] : 0.f;
        w16[OFF_DTW + i] = (_Float16)v; return;
    }
    i -= NLAYERS * DI * 32;
    if (i < NLAYERS * DM * DI) { w16[OFF_OUTW + i] = (_Float16)out_w[i]; return; }
    i -= NLAYERS * DM * DI;
    if (i < NLAYERS * DI * DSTATE) { A_ws[i] = -__expf(A_log[i]); return; }
}

// ---------------- one 16x16 WMMA tile: C += A(LDS f16) x B(global f16) ----------------
// A row-major [M][lda], op layout: lane holds row (lane&15), K halves {kb..kb+7, kb+16..kb+23}, kb=(lane>>4)*8
// B from W row-major [N][ldb] (B[k][n] = W[n][k]): lane holds col (lane&15), K halves {kb..kb+7, kb+16..kb+23}, kb=(lane>>4)*8
__device__ __forceinline__ v8f wmma_tile_f16(const _Float16* __restrict__ A, int lda,
                                             const _Float16* __restrict__ W, int ldb,
                                             int m0, int n0, int K, int lane)
{
    v8f acc = {};
    const int ar    = m0 + (lane & 15);
    const int bc    = n0 + (lane & 15);
    const int koffA = (lane >> 4) << 3;
    const int koffB = (lane >> 4) << 3;
    for (int kt = 0; kt < K; kt += 32) {
        const _Float16* ap = A + ar * lda + kt + koffA;
        v8h alo = *(const v8h*)(ap);
        v8h ahi = *(const v8h*)(ap + 16);
        const _Float16* bp = W + bc * ldb + kt + koffB;
        v8h blo = *(const v8h*)(bp);
        v8h bhi = *(const v8h*)(bp + 16);
        v16h a, b;
#pragma unroll
        for (int q = 0; q < 8; ++q) { a[q] = alo[q]; a[q + 8] = ahi[q];
                                      b[q] = blo[q]; b[q + 8] = bhi[q]; }
        acc = __builtin_amdgcn_wmma_f32_16x16x32_f16(false, a, false, b,
                                                     (short)0, acc, false, false);
    }
    return acc;
}

// ---------------- fused forward: one block per surviving sequence (b, n=0) ----------------
__global__ __launch_bounds__(256) void mamba_fused_kernel(
    const float* __restrict__ x_in,  const int* __restrict__ mask,
    const float* __restrict__ mlp_b1, const float* __restrict__ mlp_b2,
    const float* __restrict__ conv_w, const float* __restrict__ conv_b,
    const float* __restrict__ dt_b,   const float* __restrict__ Dskip,
    const float* __restrict__ blk_nw, const float* __restrict__ norm_fw,
    const _Float16* __restrict__ w16, const float* __restrict__ A_ws,
    float* __restrict__ out)
{
    extern __shared__ __align__(16) char smem[];
    _Float16* hn16  = (_Float16*)(smem + SM_HN16);
    _Float16* xc16  = (_Float16*)(smem + SM_XC16);
    _Float16* z16   = (_Float16*)(smem + SM_Z16);
    _Float16* dtin  = (_Float16*)(smem + SM_DTIN);
    float*    resid = (float*)(smem + SM_RESID);
    float*    hbuf  = (float*)(smem + SM_HBUF);
    float*    X     = (float*)(smem + SM_X);
    float*    xdbl  = (float*)(smem + SM_XDBL);
    float*    sred  = (float*)(smem + SM_SRED);

    const int tid  = threadIdx.x;
    const int lane = tid & 31;
    const int wid  = tid >> 5;
    const int b    = blockIdx.x;

    // ---- load x (sequence r = b*Nseq, all 64 tokens) as f16 ----
    const float* xseq = x_in + (size_t)b * (Nseq * Lseq * DM);   // n=0 slice
    for (int i = tid; i < Lseq * DM; i += 256) hn16[i] = (_Float16)xseq[i];
    __syncthreads();

    // ---- MLP layer 1: gelu(x @ w1.T + b1) -> xc16[64][128] ----
    {
        const _Float16* W = w16 + OFF_MLP1;
        __builtin_prefetch(W, 0, 1);
        for (int tile = wid; tile < 4 * 8; tile += 8) {
            int tm = tile & 3, tn = tile >> 2;
            v8f acc = wmma_tile_f16(hn16, DM, W, DM, tm * 16, tn * 16, DM, lane);
            int n  = tn * 16 + (lane & 15);
            int mb = tm * 16 + ((lane >> 4) << 3);
            float bias = mlp_b1[n];
#pragma unroll
            for (int r = 0; r < 8; ++r) {
                float v = acc[r] + bias;
                float g = 0.5f * v * (1.f + erff(v * 0.70710678f));
                xc16[(mb + r) * DM + n] = (_Float16)g;
            }
        }
    }
    __syncthreads();
    // ---- MLP layer 2: -> hbuf f32 ----
    {
        const _Float16* W = w16 + OFF_MLP2;
        __builtin_prefetch(W, 0, 1);
        for (int tile = wid; tile < 4 * 8; tile += 8) {
            int tm = tile & 3, tn = tile >> 2;
            v8f acc = wmma_tile_f16(xc16, DM, W, DM, tm * 16, tn * 16, DM, lane);
            int n  = tn * 16 + (lane & 15);
            int mb = tm * 16 + ((lane >> 4) << 3);
            float bias = mlp_b2[n];
#pragma unroll
            for (int r = 0; r < 8; ++r) hbuf[(mb + r) * DM + n] = acc[r] + bias;
        }
    }
    __syncthreads();

    // ---- mamba layers ----
    for (int l = 0; l < NLAYERS; ++l) {
        for (int i = tid; i < Lseq * DM; i += 256)
            resid[i] = (l == 0) ? hbuf[i] : hbuf[i] + resid[i];
        __syncthreads();
        // rmsnorm -> hn16 (one row per thread, 64 rows)
        if (tid < Lseq) {
            const float* rr = resid + tid * DM;
            float ss = 0.f;
#pragma unroll 4
            for (int f = 0; f < DM; ++f) ss += rr[f] * rr[f];
            float sc = rsqrtf(ss * (1.f / DM) + 1e-5f);
            for (int f = 0; f < DM; ++f)
                hn16[tid * DM + f] = (_Float16)(rr[f] * sc * blk_nw[l * DM + f]);
        }
        __syncthreads();
        // in_proj: (64x128)@(128->512); cols<256 -> X (xc), cols>=256 -> z16
        {
            const _Float16* W = w16 + OFF_INW + l * (2 * DI) * DM;
            __builtin_prefetch(W, 0, 1);
            for (int tile = wid; tile < 4 * 32; tile += 8) {
                int tm = tile & 3, tn = tile >> 2;
                v8f acc = wmma_tile_f16(hn16, DM, W, DM, tm * 16, tn * 16, DM, lane);
                int n  = tn * 16 + (lane & 15);
                int mb = tm * 16 + ((lane >> 4) << 3);
                if (n < DI) {
#pragma unroll
                    for (int r = 0; r < 8; ++r) X[(mb + r) * DI + n] = acc[r];
                } else {
#pragma unroll
                    for (int r = 0; r < 8; ++r) z16[(mb + r) * DI + (n - DI)] = (_Float16)acc[r];
                }
            }
        }
        __syncthreads();
        // causal conv1d + silu -> xc16 (thread = channel, sliding window over t)
        {
            const int d = tid;
            const float w0 = conv_w[(l * DI + d) * DCONV + 0];
            const float w1 = conv_w[(l * DI + d) * DCONV + 1];
            const float w2 = conv_w[(l * DI + d) * DCONV + 2];
            const float w3 = conv_w[(l * DI + d) * DCONV + 3];
            const float cb = conv_b[l * DI + d];
            float x0 = 0.f, x1 = 0.f, x2 = 0.f;
            for (int t = 0; t < Lseq; ++t) {
                float x3 = X[t * DI + d];
                float y  = cb + w0 * x0 + w1 * x1 + w2 * x2 + w3 * x3;
                y = y / (1.f + __expf(-y));
                xc16[t * DI + d] = (_Float16)y;
                x0 = x1; x1 = x2; x2 = x3;
            }
        }
        __syncthreads();
        // x_proj: (64x256)@(256->48 padded) -> xdbl
        {
            const _Float16* W = w16 + OFF_XPW + l * 48 * DI;
            __builtin_prefetch(W, 0, 1);
            for (int tile = wid; tile < 4 * 3; tile += 8) {
                int tm = tile & 3, tn = tile >> 2;
                v8f acc = wmma_tile_f16(xc16, DI, W, DI, tm * 16, tn * 16, DI, lane);
                int n  = tn * 16 + (lane & 15);
                int mb = tm * 16 + ((lane >> 4) << 3);
#pragma unroll
                for (int r = 0; r < 8; ++r) xdbl[(mb + r) * 48 + n] = acc[r];
            }
        }
        __syncthreads();
        // pack dt_rank slice into zero-padded f16 A [64][32]
        for (int i = tid; i < Lseq * 32; i += 256) dtin[i] = (_Float16)0.f;
        __syncthreads();
        for (int i = tid; i < Lseq * DTRANK; i += 256) {
            int t = i >> 3, c = i & 7;
            dtin[t * 32 + c] = (_Float16)xdbl[t * 48 + c];
        }
        __syncthreads();
        // dt_proj: (64x32)@(32->256), softplus(+dt_b) -> X
        {
            const _Float16* W = w16 + OFF_DTW + l * DI * 32;
            __builtin_prefetch(W, 0, 1);
            for (int tile = wid; tile < 4 * 16; tile += 8) {
                int tm = tile & 3, tn = tile >> 2;
                v8f acc = wmma_tile_f16(dtin, 32, W, 32, tm * 16, tn * 16, 32, lane);
                int n  = tn * 16 + (lane & 15);
                int mb = tm * 16 + ((lane >> 4) << 3);
                float bias = dt_b[l * DI + n];
#pragma unroll
                for (int r = 0; r < 8; ++r) {
                    float v  = acc[r] + bias;
                    float sp = (v > 20.f) ? v : log1pf(__expf(v));
                    X[(mb + r) * DI + n] = sp;
                }
            }
        }
        __syncthreads();
        // selective scan: thread = channel d; 16 states in registers; B/C broadcast from LDS
        {
            const int d = tid;
            float Ar[DSTATE], hs[DSTATE];
            const float* Ap = A_ws + (l * DI + d) * DSTATE;
#pragma unroll
            for (int n = 0; n < DSTATE; ++n) { Ar[n] = Ap[n]; hs[n] = 0.f; }
            const float Dsk = Dskip[l * DI + d];
            for (int t = 0; t < Lseq; ++t) {
                float dtv = X[t * DI + d];
                float u   = (float)xc16[t * DI + d];
                float du  = dtv * u;
                const float* Bt = xdbl + t * 48 + DTRANK;
                const float* Ct = xdbl + t * 48 + DTRANK + DSTATE;
                float y = 0.f;
#pragma unroll
                for (int n = 0; n < DSTATE; ++n) {
                    float dA = __expf(dtv * Ar[n]);
                    hs[n] = dA * hs[n] + du * Bt[n];
                    y += hs[n] * Ct[n];
                }
                X[t * DI + d] = y + Dsk * u;   // overwrite dt with y in place
            }
        }
        __syncthreads();
        // gate: xc16 = f16( y * silu(z) )
        for (int i = tid; i < Lseq * DI; i += 256) {
            float zz = (float)z16[i];
            float g  = zz / (1.f + __expf(-zz));
            xc16[i]  = (_Float16)(X[i] * g);
        }
        __syncthreads();
        // out_proj: (64x256)@(256->128) -> hbuf
        {
            const _Float16* W = w16 + OFF_OUTW + l * DM * DI;
            __builtin_prefetch(W, 0, 1);
            for (int tile = wid; tile < 4 * 8; tile += 8) {
                int tm = tile & 3, tn = tile >> 2;
                v8f acc = wmma_tile_f16(xc16, DI, W, DI, tm * 16, tn * 16, DI, lane);
                int n  = tn * 16 + (lane & 15);
                int mb = tm * 16 + ((lane >> 4) << 3);
#pragma unroll
                for (int r = 0; r < 8; ++r) hbuf[(mb + r) * DM + n] = acc[r];
            }
        }
        __syncthreads();
    }

    // ---- final residual + rmsnorm(row 63) + key_valid mask ----
    for (int i = tid; i < Lseq * DM; i += 256) resid[i] += hbuf[i];
    __syncthreads();
    if (tid == 0) {
        const float* rr = resid + (Lseq - 1) * DM;
        float ss = 0.f;
        for (int f = 0; f < DM; ++f) ss += rr[f] * rr[f];
        sred[0] = rsqrtf(ss * (1.f / DM) + 1e-5f);
        int anyv = 0;
        const int* mrow = mask + b * (Nseq * Lseq);   // row r = b*Nseq, n=0
        for (int t = 0; t < Lseq; ++t) anyv |= (mrow[t] != 0);
        sred[1] = anyv ? 1.f : 0.f;
    }
    __syncthreads();
    if (tid < DM)
        out[b * DM + tid] = resid[(Lseq - 1) * DM + tid] * sred[0] * norm_fw[tid] * sred[1];
}

// ---------------- launch ----------------
extern "C" void kernel_launch(void* const* d_in, const int* in_sizes, int n_in,
                              void* d_out, int out_size, void* d_ws, size_t ws_size,
                              hipStream_t stream) {
    const float* mha_in  = (const float*)d_in[0];
    const int*   mask    = (const int*)  d_in[1];   // bernoulli bool -> integer
    const float* mlp_w1  = (const float*)d_in[2];
    const float* mlp_b1  = (const float*)d_in[3];
    const float* mlp_w2  = (const float*)d_in[4];
    const float* mlp_b2  = (const float*)d_in[5];
    const float* in_w    = (const float*)d_in[6];
    const float* conv_w  = (const float*)d_in[7];
    const float* conv_b  = (const float*)d_in[8];
    const float* xp_w    = (const float*)d_in[9];
    const float* dt_w    = (const float*)d_in[10];
    const float* dt_b    = (const float*)d_in[11];
    const float* A_log   = (const float*)d_in[12];
    const float* Dskip   = (const float*)d_in[13];
    const float* out_w   = (const float*)d_in[14];
    const float* blk_nw  = (const float*)d_in[15];
    const float* norm_fw = (const float*)d_in[16];

    _Float16* w16  = (_Float16*)d_ws;
    float*    A_ws = (float*)((char*)d_ws + (size_t)TOTAL_H * 2);
    float*    out  = (float*)d_out;

    const int prep_total = TOTAL_H + NLAYERS * DI * DSTATE;
    mamba_prep_kernel<<<(prep_total + 255) / 256, 256, 0, stream>>>(
        mlp_w1, mlp_w2, in_w, xp_w, dt_w, out_w, A_log, w16, A_ws);

    mamba_fused_kernel<<<Bsz, 256, SMEM_BYTES, stream>>>(
        mha_in, mask, mlp_b1, mlp_b2, conv_w, conv_b, dt_b, Dskip,
        blk_nw, norm_fw, w16, A_ws, out);
}
